// Seq2Seq_72627896975362
// MI455X (gfx1250) — matrix operand order repaired and checked by reading, not compiled, not run on hardware
//
#include <hip/hip_runtime.h>
#include <hip/hip_bf16.h>
#include <stdint.h>

// ---------------------------------------------------------------------------
// Seq2Seq LSTM for MI455X (gfx1250, wave32, WMMA).
//   V=32000 H=512 L=3 S=128 T=128  -> out [129, 32000] f32
//
// Pipeline (all on `stream`, deterministic, no float atomics):
//   1) init_kernel        : zero barriers/states, seed dec row0 (one-hot sos),
//                           convert recurrent weights f32->bf16 into ws (21MB).
//   2) input_gemm_kernel  : Zpart[k] = A_tile @ W0[:V,:]  (bf16 WMMA, K-split)
//   3) reduce_kernel      : Zx = bias + sum_k Zpart[k]    (fixed order)
//   4) recurrent_kernel   : persistent cooperative kernel, layer-wavefront
//                           pipelined LSTM, 1 grid barrier per tick.
//   5) logits_gemm_kernel : out = Hd @ Wf + bf            (bf16 WMMA)
// ---------------------------------------------------------------------------

typedef __attribute__((ext_vector_type(16))) __bf16 v16bf;
typedef __attribute__((ext_vector_type(8)))  float  v8f;

#define V_DIM   32000
#define H_DIM   512
#define GATE4   2048
#define NWG_REC 48
#define KCHUNKS 10
#define BARR_SLOTS 384

// ---- workspace layout (bytes) ----
#define WS_BARR  0                                 // 384 * u32
#define WS_ZXE   4096                              // 128*2048 f32
#define WS_ZXD   (WS_ZXE + 128*2048*4)             // 129*2048 f32
#define WS_HD    (WS_ZXD + 129*2048*4)             // 144*512 f32 (padded rows)
#define WS_ABUF  (WS_HD + 144*512*4)               // 3 layers * 2 parity * 512
#define WS_CBUF  (WS_ABUF + 3*2*512*4)
#define WS_WB    (WS_CBUF + 3*2*512*4)             // 10,485,760 bf16 (recurrent W)
#define WS_ZPE   (WS_WB + 10485760*2)              // 10*128*2048 f32 partials
#define WS_ZPD   (WS_ZPE + KCHUNKS*128*2048*4)     // 10*128*2048 f32 partials
// total ~44.4 MB

__device__ __forceinline__ float sigf(float x) { return 1.f / (1.f + __expf(-x)); }
__device__ __forceinline__ float tanhfast(float x) { return 2.f / (1.f + __expf(-2.f * x)) - 1.f; }

__device__ __forceinline__ v8f wmma_bf16(v16bf a, v16bf b, v8f c) {
    // emits v_wmma_f32_16x16x32_bf16 (args: neg_a, A, neg_b, B, c_mod, C, reuse_a, reuse_b)
    return __builtin_amdgcn_wmma_f32_16x16x32_bf16(false, a, false, b, (short)0, c, false, false);
}

// ---------------------------------------------------------------------------
// init: barriers, states, dec row0 = W0[sos,:] + b0, recurrent weights -> bf16
// ---------------------------------------------------------------------------
__global__ void init_kernel(const float* __restrict__ dec_b0,
                            const float* __restrict__ enc_W0,
                            const float* __restrict__ dec_W0,
                            const float* __restrict__ enc_W,
                            const float* __restrict__ dec_W,
                            const int* __restrict__ sosp,
                            unsigned* __restrict__ barr,
                            float* __restrict__ Zxd,
                            float* __restrict__ Hd,
                            float* __restrict__ Ab,
                            float* __restrict__ Cb,
                            __bf16* __restrict__ Wb)
{
    const int gid = blockIdx.x * blockDim.x + threadIdx.x;
    const int stride = gridDim.x * blockDim.x;

    for (int i = gid; i < BARR_SLOTS; i += stride) barr[i] = 0u;
    for (int i = gid; i < 144 * 512; i += stride) Hd[i] = 0.f;
    for (int i = gid; i < 3 * 2 * 512; i += stride) { Ab[i] = 0.f; Cb[i] = 0.f; }

    const int sos = sosp[0];
    for (int i = gid; i < GATE4; i += stride)            // dec layer0 step0: one-hot(sos)
        Zxd[i] = dec_b0[i] + dec_W0[(size_t)sos * GATE4 + i];

    const size_t N0 = (size_t)512 * GATE4;               // layer0 recurrent slab
    const size_t N12 = (size_t)2 * 1024 * GATE4;         // layers 1..2 (contiguous)
    { const float* s = enc_W0 + (size_t)V_DIM * GATE4; __bf16* d = Wb;
      for (size_t i = gid; i < N0;  i += stride) d[i] = (__bf16)s[i]; }
    { const float* s = enc_W;                          __bf16* d = Wb + 1048576;
      for (size_t i = gid; i < N12; i += stride) d[i] = (__bf16)s[i]; }
    { const float* s = dec_W0 + (size_t)V_DIM * GATE4; __bf16* d = Wb + 5242880;
      for (size_t i = gid; i < N0;  i += stride) d[i] = (__bf16)s[i]; }
    { const float* s = dec_W;                          __bf16* d = Wb + 6291456;
      for (size_t i = gid; i < N12; i += stride) d[i] = (__bf16)s[i]; }
}

// ---------------------------------------------------------------------------
// input GEMM: Zpart[kchunk][128][2048] = A[128,32000] @ W[:32000, 2048]
// one wave per (kchunk, 128-col block, 16-row tile); 8 WMMA tiles per wave
// ---------------------------------------------------------------------------
__global__ __launch_bounds__(32) void input_gemm_kernel(const float* __restrict__ A,
                                                        const float* __restrict__ W,
                                                        float* __restrict__ Zp)
{
    const int lane = threadIdx.x;
    const int kb0  = blockIdx.x * (V_DIM / KCHUNKS);   // 3200-wide K chunk
    const int nb   = blockIdx.y * 128;
    const int mt   = blockIdx.z * 16;

    v8f acc[8] = {};
    const int m    = mt + (lane & 15);
    const int koff = (lane >> 4) << 3;                 // A frag: K sub-offset per half-wave
    const int nlo  = lane & 15;
    const int khi  = (lane >> 4) << 3;                 // B frag: K sub-offset per half-wave

    for (int kb = kb0; kb < kb0 + (V_DIM / KCHUNKS); kb += 32) {
        v16bf a;
        const float* pa = A + (size_t)m * V_DIM + kb + koff;
#pragma unroll
        for (int e = 0; e < 8; ++e) { a[e] = (__bf16)pa[e]; a[8 + e] = (__bf16)pa[16 + e]; }
#pragma unroll
        for (int s = 0; s < 8; ++s) {
            const float* pb = W + (size_t)(kb + khi) * GATE4 + nb + s * 16 + nlo;
            v16bf b;
#pragma unroll
            for (int e = 0; e < 8; ++e) { b[e] = (__bf16)pb[(size_t)e * GATE4]; b[8 + e] = (__bf16)pb[(size_t)(16 + e) * GATE4]; }
            acc[s] = wmma_bf16(a, b, acc[s]);
        }
    }
    float* Zslab = Zp + (size_t)blockIdx.x * 128 * GATE4;
    const int rbase = mt + ((lane >> 4) << 3);
#pragma unroll
    for (int s = 0; s < 8; ++s) {
        const int col = nb + s * 16 + nlo;
#pragma unroll
        for (int i = 0; i < 8; ++i)
            Zslab[(size_t)(rbase + i) * GATE4 + col] = acc[s][i];
    }
}

// deterministic fixed-order reduction of K-split partials + bias
__global__ void reduce_kernel(const float* __restrict__ Zp,
                              const float* __restrict__ bias,
                              float* __restrict__ dst)
{
    const int i = blockIdx.x * blockDim.x + threadIdx.x;
    if (i >= 128 * GATE4) return;
    float s = bias[i & (GATE4 - 1)];
#pragma unroll
    for (int k = 0; k < KCHUNKS; ++k) s += Zp[(size_t)k * 128 * GATE4 + i];
    dst[i] = s;
}

// ---------------------------------------------------------------------------
// persistent cooperative recurrence, layer-wavefront pipelined:
// tick tau: layer l computes its step t = tau - l.  1 grid barrier / tick.
// 48 WGs x 256: groups of 8 WGs per (stack, layer) matrix.
// a/c state double-buffered by step parity -> no intra-tick hazards.
// ---------------------------------------------------------------------------
__global__ __launch_bounds__(256) void recurrent_kernel(const float* __restrict__ Zxe,
                                                        const float* __restrict__ Zxd,
                                                        float* __restrict__ Ab,
                                                        float* __restrict__ Cb,
                                                        float* __restrict__ Hd,
                                                        const __bf16* __restrict__ Wb,
                                                        const float* __restrict__ enc_b,
                                                        const float* __restrict__ dec_b,
                                                        volatile unsigned* barr)
{
    const int wg  = blockIdx.x;
    const int grp = wg >> 3;         // 0..5 : (enc|dec) x layer
    const int k8  = wg & 7;          // wg index within group (owns 64 hidden units)
    const int myStack = grp / 3;
    const int l   = grp % 3;
    const int tid = threadIdx.x;

    __shared__ float hb_s[512];
    __shared__ float ap_s[512];
    __shared__ float z_s[256];

    const size_t wbase = (size_t)myStack * 5242880u
                       + (l == 0 ? 0u : (l == 1 ? 1048576u : 3145728u));

    int btick = 0;
    for (int phase = 0; phase < 2; ++phase) {
        const int steps = (phase == 0) ? 128 : 129;
        const float* Zx   = (phase == 0) ? Zxe : Zxd;
        const float* brec = (phase == 0) ? enc_b : dec_b;
        const bool activeStack = (myStack == phase);

        for (int tk = 0; tk < steps + 2; ++tk, ++btick) {
            const int t = tk - l;
            if (activeStack && t >= 0 && t < steps) {
                const int pprev = (t - 1) & 1;
                const int pcur  = t & 1;
                // stage a_l(t-1) and h_{l-1}(t) into LDS (broadcast reads later)
                const float* apg = Ab + (size_t)(l * 2 + pprev) * 512;
                for (int r = tid; r < 512; r += 256) ap_s[r] = apg[r];
                if (l > 0) {
                    const float* hbg = Ab + (size_t)((l - 1) * 2 + pcur) * 512;
                    for (int r = tid; r < 512; r += 256) hb_s[r] = hbg[r];
                }
                __syncthreads();

                const int gi = tid >> 6;                 // gate: f,u,g,o
                const int jl = tid & 63;
                const int c  = gi * 512 + k8 * 64 + jl;  // gate column, coalesced
                const __bf16* Wc = Wb + wbase + c;
                float z;
                if (l == 0) {
                    z = Zx[(size_t)t * GATE4 + c];       // precomputed x@W + b
#pragma unroll 8
                    for (int r = 0; r < 512; ++r) z += ap_s[r] * (float)Wc[(size_t)r * GATE4];
                } else {
                    z = brec[(size_t)(l - 1) * GATE4 + c];
#pragma unroll 8
                    for (int r = 0; r < 512; ++r) z += hb_s[r] * (float)Wc[(size_t)r * GATE4];
                    const __bf16* Wc2 = Wc + (size_t)512 * GATE4;
#pragma unroll 8
                    for (int r = 0; r < 512; ++r) z += ap_s[r] * (float)Wc2[(size_t)r * GATE4];
                }
                z_s[tid] = z;
                __syncthreads();

                if (tid < 64) {                          // elementwise LSTM update
                    const int jj = k8 * 64 + tid;
                    const float f = z_s[tid], u = z_s[64 + tid];
                    const float g = z_s[128 + tid], o = z_s[192 + tid];
                    const float cp = Cb[(size_t)(l * 2 + pprev) * 512 + jj];
                    const float cn = sigf(f) * cp + sigf(u) * tanhfast(g);
                    const float an = tanhfast(cn) * sigf(o);
                    Ab[(size_t)(l * 2 + pcur) * 512 + jj] = an;
                    Cb[(size_t)(l * 2 + pcur) * 512 + jj] = cn;
                    if (phase == 1 && l == 2) Hd[(size_t)t * 512 + jj] = an;
                    __threadfence();
                }
                __syncthreads();
            }
            // grid barrier: one fresh counter per tick (re-zeroed by init each launch)
            if (tid == 0) {
                __threadfence();
                atomicAdd((unsigned*)(barr + btick), 1u);
                while (barr[btick] < (unsigned)NWG_REC) __builtin_amdgcn_s_sleep(2);
            }
            __syncthreads();
        }
    }
}

// ---------------------------------------------------------------------------
// logits: out[129,32000] = Hd[129,512] @ Wf[512,32000] + bf   (bf16 WMMA)
// ---------------------------------------------------------------------------
__global__ __launch_bounds__(32) void logits_gemm_kernel(const float* __restrict__ Hd,
                                                         const float* __restrict__ Wf,
                                                         const float* __restrict__ bf,
                                                         float* __restrict__ out)
{
    const int lane = threadIdx.x;
    const int nb = blockIdx.x * 128;
    const int mt = blockIdx.y * 16;

    v8f acc[8] = {};
    const int m    = mt + (lane & 15);
    const int koff = (lane >> 4) << 3;
    const int nlo  = lane & 15;
    const int khi  = (lane >> 4) << 3;

    for (int kb = 0; kb < H_DIM; kb += 32) {
        v16bf a;
        const float* pa = Hd + (size_t)m * H_DIM + kb + koff;
#pragma unroll
        for (int e = 0; e < 8; ++e) { a[e] = (__bf16)pa[e]; a[8 + e] = (__bf16)pa[16 + e]; }
#pragma unroll
        for (int s = 0; s < 8; ++s) {
            const float* pb = Wf + (size_t)(kb + khi) * V_DIM + nb + s * 16 + nlo;
            v16bf b;
#pragma unroll
            for (int e = 0; e < 8; ++e) { b[e] = (__bf16)pb[(size_t)e * V_DIM]; b[8 + e] = (__bf16)pb[(size_t)(16 + e) * V_DIM]; }
            acc[s] = wmma_bf16(a, b, acc[s]);
        }
    }
    const int rbase = mt + ((lane >> 4) << 3);
#pragma unroll
    for (int s = 0; s < 8; ++s) {
        const int col = nb + s * 16 + nlo;
        const float bias = bf[col];
#pragma unroll
        for (int i = 0; i < 8; ++i) {
            const int row = rbase + i;
            if (row < 129) out[(size_t)row * V_DIM + col] = acc[s][i] + bias;
        }
    }
}

extern "C" void kernel_launch(void* const* d_in, const int* in_sizes, int n_in,
                              void* d_out, int out_size, void* d_ws, size_t ws_size,
                              hipStream_t stream)
{
    const float* seq    = (const float*)d_in[0];
    const float* tgt    = (const float*)d_in[1];
    const float* enc_W0 = (const float*)d_in[2];
    const float* enc_b0 = (const float*)d_in[3];
    const float* enc_W  = (const float*)d_in[4];
    const float* enc_b  = (const float*)d_in[5];
    const float* dec_W0 = (const float*)d_in[6];
    const float* dec_b0 = (const float*)d_in[7];
    const float* dec_W  = (const float*)d_in[8];
    const float* dec_b  = (const float*)d_in[9];
    const float* dec_Wf = (const float*)d_in[10];
    const float* dec_bf = (const float*)d_in[11];
    const int*   sosp   = (const int*)d_in[12];
    float* out = (float*)d_out;

    char* ws = (char*)d_ws;
    unsigned* barr = (unsigned*)(ws + WS_BARR);
    float* Zxe = (float*)(ws + WS_ZXE);
    float* Zxd = (float*)(ws + WS_ZXD);
    float* Hd  = (float*)(ws + WS_HD);
    float* Ab  = (float*)(ws + WS_ABUF);
    float* Cb  = (float*)(ws + WS_CBUF);
    __bf16* Wb = (__bf16*)(ws + WS_WB);
    float* Zpe = (float*)(ws + WS_ZPE);
    float* Zpd = (float*)(ws + WS_ZPD);

    init_kernel<<<2048, 256, 0, stream>>>(dec_b0, enc_W0, dec_W0, enc_W, dec_W, sosp,
                                          barr, Zxd, Hd, Ab, Cb, Wb);

    input_gemm_kernel<<<dim3(KCHUNKS, 16, 8), 32, 0, stream>>>(seq, enc_W0, Zpe);
    input_gemm_kernel<<<dim3(KCHUNKS, 16, 8), 32, 0, stream>>>(tgt, dec_W0, Zpd);

    reduce_kernel<<<(128 * GATE4 + 255) / 256, 256, 0, stream>>>(Zpe, enc_b0, Zxe);
    reduce_kernel<<<(128 * GATE4 + 255) / 256, 256, 0, stream>>>(Zpd, dec_b0, Zxd + GATE4);

    recurrent_kernel<<<NWG_REC, 256, 0, stream>>>(Zxe, Zxd, Ab, Cb, Hd, Wb,
                                                  enc_b, dec_b, barr);

    logits_gemm_kernel<<<dim3(250, 9), 32, 0, stream>>>(Hd, dec_Wf, dec_bf, out);
}
